// MultiHeadAttention_47734266528380
// MI455X (gfx1250) — compile-verified, matrix operand order checked by reading, not run on hardware
//
#include <hip/hip_runtime.h>
#include <hip/hip_bf16.h>
#include <stdint.h>

// Problem constants (from reference): B=2, S=2048, E=1024, H=16, Dh=64
#define BATCH 2
#define SEQ   2048
#define EMB   1024
#define NH    16
#define DH    64

typedef __bf16 bf16_t;
typedef bf16_t bf16x16 __attribute__((ext_vector_type(16)));
typedef float  floatx8 __attribute__((ext_vector_type(8)));

union ABFrag {
  bf16x16 v;
  uint4   q[2];
  unsigned short s[16];
};

__device__ __forceinline__ unsigned short f2bf(float f) {
  union { float f; unsigned int u; } c; c.f = f;
  unsigned int u = c.u;
  unsigned int r = u + 0x7FFFu + ((u >> 16) & 1u);  // round-to-nearest-even
  return (unsigned short)(r >> 16);
}

// ---------------------------------------------------------------------------
// Async global->LDS staging (CDNA5 GLOBAL_LOAD_ASYNC_TO_LDS_B128, ASYNCcnt).
// Builtin prototype (from hipcc diagnostic): param0 = int4 ptr in AS1
// (printed as "__device__"), so pass fully-typed vector pointers.
// ---------------------------------------------------------------------------
typedef int i32x4 __attribute__((vector_size(16)));
#define GAS __attribute__((address_space(1)))
#define LAS __attribute__((address_space(3)))

#if defined(__HIP_DEVICE_COMPILE__) && __has_builtin(__builtin_amdgcn_global_load_async_to_lds_b128)
#define ASYNC_LDS 1
#else
#define ASYNC_LDS 0
#endif

__device__ __forceinline__ void cp16_to_lds(void* ldst, const void* gsrc) {
#if ASYNC_LDS
  // generic->AS1: global generic addresses are identity.
  // generic->AS3: flat LDS aperture keeps LDS byte addr in addr[31:0].
  __builtin_amdgcn_global_load_async_to_lds_b128(
      (GAS i32x4*)(uintptr_t)gsrc,
      (LAS i32x4*)(uint32_t)(uintptr_t)ldst,
      0, 0);
#else
  *(uint4*)ldst = *(const uint4*)gsrc;
#endif
}

__device__ __forceinline__ void async_wait_all() {
#if ASYNC_LDS
#if __has_builtin(__builtin_amdgcn_s_wait_asynccnt)
  __builtin_amdgcn_s_wait_asynccnt(0);
#else
  asm volatile("s_wait_asynccnt 0" ::: "memory");
#endif
#endif
}

// ---------------------------------------------------------------------------
// Kernel 0: convert Wo f32 -> bf16 once
// ---------------------------------------------------------------------------
__global__ void mha_cvt_kernel(const float* __restrict__ src,
                               unsigned short* __restrict__ dst, int n) {
  int i = blockIdx.x * blockDim.x + threadIdx.x;
  int stride = gridDim.x * blockDim.x;
  for (; i < n; i += stride) dst[i] = f2bf(src[i]);
}

// ---------------------------------------------------------------------------
// Kernel 1: per-head QKV projection.
//   grid = (B*H, S/64), block = 128 (4 waves). Wave w owns 16 query rows.
//   Q, K stored row-major bf16 [bh][s][d]; V stored transposed [bh][d][s].
// ---------------------------------------------------------------------------
__global__ __launch_bounds__(128) void mha_qkv_kernel(
    const float* __restrict__ x,
    const float* __restrict__ Wq, const float* __restrict__ Wk,
    const float* __restrict__ Wv,
    const float* __restrict__ bq, const float* __restrict__ bk,
    const float* __restrict__ bv,
    unsigned short* __restrict__ Qb, unsigned short* __restrict__ Kb,
    unsigned short* __restrict__ Vtb) {
  __shared__ __align__(16) unsigned short sW[3][DH * DH];  // transposed [n][k]
  __shared__ __align__(16) unsigned short sX[64 * DH];     // [row][k]

  const int bh = blockIdx.x;
  const int b = bh / NH, h = bh % NH;
  const int q0 = blockIdx.y * 64;
  const int tid = threadIdx.x;
  const int lane = tid & 31;
  const int wv = tid >> 5;
  const int hi = lane >> 4;
  const int lm = lane & 15;

  const float* Wsrc[3] = { Wq + (size_t)h * DH * DH,
                           Wk + (size_t)h * DH * DH,
                           Wv + (size_t)h * DH * DH };
#pragma unroll
  for (int m = 0; m < 3; ++m) {
    const float* Wm = Wsrc[m];
    for (int j = 0; j < 32; ++j) {
      int f = tid + 128 * j;            // 0..4095
      int k = f >> 6, n = f & 63;
      sW[m][n * DH + k] = f2bf(Wm[f]);  // transpose into [n][k]
    }
  }
  const float* xrow = x + ((size_t)b * SEQ + q0) * EMB + h * DH;
  for (int j = 0; j < 32; ++j) {
    int f = tid + 128 * j;
    int r = f >> 6, k = f & 63;
    sX[r * DH + k] = f2bf(xrow[(size_t)r * EMB + k]);
  }
  __syncthreads();

  // A fragments: 16x64 strip of X, two K=32 steps.
  const int arow = wv * 16 + lm;
  ABFrag a[2];
#pragma unroll
  for (int s = 0; s < 2; ++s) {
    const uint4* p = (const uint4*)&sX[arow * DH + s * 32 + hi * 8];
    a[s].q[0] = p[0];
    a[s].q[1] = p[2];  // +16 elements = +2 uint4
  }

  const float* bias[3] = { bq + h * DH, bk + h * DH, bv + h * DH };
  const size_t base = (size_t)bh * SEQ * DH;
#pragma unroll
  for (int m = 0; m < 3; ++m) {
#pragma unroll
    for (int t = 0; t < 4; ++t) {
      floatx8 c;
      float bval = bias[m][t * 16 + lm];
#pragma unroll
      for (int r = 0; r < 8; ++r) c[r] = bval;
#pragma unroll
      for (int s = 0; s < 2; ++s) {
        ABFrag bf;
        const uint4* p = (const uint4*)&sW[m][(t * 16 + lm) * DH + s * 32 + hi * 8];
        bf.q[0] = p[0]; bf.q[1] = p[2];
        c = __builtin_amdgcn_wmma_f32_16x16x32_bf16(
                false, a[s].v, false, bf.v, (short)0, c, false, false);
      }
      if (m < 2) {
        unsigned short* dst = (m == 0 ? Qb : Kb) + base;
#pragma unroll
        for (int r = 0; r < 8; ++r) {
          int rowg = q0 + wv * 16 + r + 8 * hi;
          dst[(size_t)rowg * DH + t * 16 + lm] = f2bf(c[r]);
        }
      } else {
#pragma unroll
        for (int r = 0; r < 8; ++r) {
          int rowg = q0 + wv * 16 + r + 8 * hi;
          Vtb[base + (size_t)(t * 16 + lm) * SEQ + rowg] = f2bf(c[r]);
        }
      }
    }
  }
}

// ---------------------------------------------------------------------------
// Kernel 2: flash attention, double-buffered async K/V tile staging.
//   grid = (B*H, S/64), block = 128 (4 waves). Wave owns a 16-row Q strip.
// ---------------------------------------------------------------------------
__device__ __forceinline__ void stage_kv(
    const unsigned short* __restrict__ Kb,
    const unsigned short* __restrict__ Vtb,
    size_t base, int kb,
    unsigned short* sKbuf, unsigned short* sVbuf, int tid) {
  // K tile: rows kb..kb+31 x 64 contiguous = 256 uint4
  const uint4* gk = (const uint4*)&Kb[base + (size_t)kb * DH];
  uint4* lk = (uint4*)sKbuf;
  cp16_to_lds(&lk[tid], &gk[tid]);
  cp16_to_lds(&lk[tid + 128], &gk[tid + 128]);
  // V^T tile: [d][32], 4 uint4 per d-row
  uint4* lv = (uint4*)sVbuf;
#pragma unroll
  for (int j = 0; j < 2; ++j) {
    int c = tid + 128 * j;
    int d = c >> 2, c8 = c & 3;
    cp16_to_lds(&lv[c], &Vtb[base + (size_t)d * SEQ + kb + c8 * 8]);
  }
}

__global__ __launch_bounds__(128) void mha_attn_kernel(
    const unsigned short* __restrict__ Qb,
    const unsigned short* __restrict__ Kb,
    const unsigned short* __restrict__ Vtb,
    unsigned short* __restrict__ valsb) {
  __shared__ __align__(16) unsigned short sK[2][32 * DH];  // [buf][krow][d]
  __shared__ __align__(16) unsigned short sV[2][DH * 32];  // [buf][d][krow]
  __shared__ __align__(16) unsigned short sP[4][16 * 32];  // per-wave P

  const int bh = blockIdx.x;
  const int b = bh / NH, h = bh % NH;
  const int q0 = blockIdx.y * 64;
  const int tid = threadIdx.x;
  const int lane = tid & 31;
  const int wv = tid >> 5;
  const int hi = lane >> 4;
  const int lm = lane & 15;
  const size_t base = (size_t)bh * SEQ * DH;

  // Q strip in A layout (kept in registers for the whole pass)
  const int qrow = q0 + wv * 16 + lm;
  ABFrag qa[2];
#pragma unroll
  for (int s = 0; s < 2; ++s) {
    const uint4* p = (const uint4*)&Qb[base + (size_t)qrow * DH + s * 32 + hi * 8];
    qa[s].q[0] = p[0]; qa[s].q[1] = p[2];
  }

  floatx8 o[4];
#pragma unroll
  for (int t = 0; t < 4; ++t)
#pragma unroll
    for (int r = 0; r < 8; ++r) o[t][r] = 0.f;
  float mrow[8], lrow[8];
#pragma unroll
  for (int r = 0; r < 8; ++r) { mrow[r] = -1e30f; lrow[r] = 0.f; }
  const float scale = 0.125f;  // 1/sqrt(64)

  // preload first tile
  stage_kv(Kb, Vtb, base, 0, sK[0], sV[0], tid);
  async_wait_all();
  __syncthreads();

  int cur = 0;
  for (int kb = 0; kb < SEQ; kb += 32) {
    // issue async loads for the next tile before computing on this one
    if (kb + 32 < SEQ)
      stage_kv(Kb, Vtb, base, kb + 32, sK[cur ^ 1], sV[cur ^ 1], tid);
    if (kb + 64 < SEQ)  // L2 prefetch two tiles ahead -> global_prefetch_b8
      __builtin_prefetch(&Kb[base + (size_t)(kb + 64) * DH + lane * 8], 0, 0);

    const unsigned short* sKc = sK[cur];
    const unsigned short* sVc = sV[cur];

    // scores = Q (16x64) x K^T (64x32) -> two 16x16 f32 tiles
    floatx8 sc[2];
#pragma unroll
    for (int t = 0; t < 2; ++t) {
      floatx8 c;
#pragma unroll
      for (int r = 0; r < 8; ++r) c[r] = 0.f;
#pragma unroll
      for (int s = 0; s < 2; ++s) {
        ABFrag kf;
        const uint4* p = (const uint4*)&sKc[(t * 16 + lm) * DH + s * 32 + hi * 8];
        kf.q[0] = p[0]; kf.q[1] = p[2];
        c = __builtin_amdgcn_wmma_f32_16x16x32_bf16(
                false, qa[s].v, false, kf.v, (short)0, c, false, false);
      }
      sc[t] = c;
    }

    // online softmax (rows live per (VGPR r, lane-half); reduce over 16 lanes)
    float alpha[8];
#pragma unroll
    for (int r = 0; r < 8; ++r) {
      float s0 = sc[0][r] * scale, s1 = sc[1][r] * scale;
      float m_ = fmaxf(s0, s1);
#pragma unroll
      for (int msk = 1; msk < 16; msk <<= 1) m_ = fmaxf(m_, __shfl_xor(m_, msk, 32));
      float mn = fmaxf(mrow[r], m_);
      float p0 = __expf(s0 - mn), p1 = __expf(s1 - mn);
      alpha[r] = __expf(mrow[r] - mn);
      float ps = p0 + p1;
#pragma unroll
      for (int msk = 1; msk < 16; msk <<= 1) ps += __shfl_xor(ps, msk, 32);
      lrow[r] = lrow[r] * alpha[r] + ps;
      mrow[r] = mn;
      sc[0][r] = p0; sc[1][r] = p1;
    }
#pragma unroll
    for (int t = 0; t < 4; ++t)
#pragma unroll
      for (int r = 0; r < 8; ++r) o[t][r] *= alpha[r];

    // C-layout -> A-layout via per-wave LDS bounce (in-wave DS ordering)
    unsigned short* pw = sP[wv];
#pragma unroll
    for (int t = 0; t < 2; ++t)
#pragma unroll
      for (int r = 0; r < 8; ++r)
        pw[(r + 8 * hi) * 32 + t * 16 + lm] = f2bf(sc[t][r]);

    ABFrag pa;
    {
      const uint4* p = (const uint4*)&pw[lm * 32 + hi * 8];
      pa.q[0] = p[0]; pa.q[1] = p[2];
    }
    // O += P (16x32) x V (32x64)
#pragma unroll
    for (int t = 0; t < 4; ++t) {
      ABFrag vf;
      const uint4* p = (const uint4*)&sVc[(t * 16 + lm) * 32 + hi * 8];
      vf.q[0] = p[0]; vf.q[1] = p[2];
      o[t] = __builtin_amdgcn_wmma_f32_16x16x32_bf16(
                 false, pa.v, false, vf.v, (short)0, o[t], false, false);
    }

    async_wait_all();   // next tile fully in LDS
    __syncthreads();    // all waves done with 'cur', next tile visible
    cur ^= 1;
  }

  // finalize: divide by row sums, write [b][s][h*64+d] bf16
#pragma unroll
  for (int t = 0; t < 4; ++t)
#pragma unroll
    for (int r = 0; r < 8; ++r) {
      int rowg = q0 + wv * 16 + r + 8 * hi;
      float val = o[t][r] / lrow[r];
      valsb[((size_t)b * SEQ + rowg) * EMB + h * DH + t * 16 + lm] = f2bf(val);
    }
}

// ---------------------------------------------------------------------------
// Kernel 3: output projection  out = vals(4096x1024) x Wo(1024x1024) + bo
//   grid = (BS/128, E/64), block = 256 (8 waves). Wave owns 16-row strip.
//   A-tile staged with async global->LDS; B-tile transposed via VGPRs.
// ---------------------------------------------------------------------------
__global__ __launch_bounds__(256) void mha_oproj_kernel(
    const unsigned short* __restrict__ valsb,
    const unsigned short* __restrict__ Wob,
    const float* __restrict__ bo, float* __restrict__ out) {
  __shared__ __align__(16) unsigned short sA[128 * 32];  // [row][k]
  __shared__ __align__(16) unsigned short sB[64 * 32];   // transposed [n][k]

  const int rb = blockIdx.x * 128;
  const int nb = blockIdx.y * 64;
  const int tid = threadIdx.x;
  const int lane = tid & 31;
  const int wv = tid >> 5;
  const int hi = lane >> 4;
  const int lm = lane & 15;

  floatx8 c[4];
#pragma unroll
  for (int t = 0; t < 4; ++t) {
    float bval = bo[nb + t * 16 + lm];
#pragma unroll
    for (int r = 0; r < 8; ++r) c[t][r] = bval;
  }

  for (int kc = 0; kc < EMB; kc += 32) {
    __syncthreads();
    {  // A tile: 128x32 = 512 uint4, 2 per thread (async global->LDS)
      uint4* la = (uint4*)sA;
#pragma unroll
      for (int j = 0; j < 2; ++j) {
        int u = tid + 256 * j;
        int r = u >> 2, c8 = u & 3;
        cp16_to_lds(&la[u], &valsb[((size_t)(rb + r)) * EMB + kc + c8 * 8]);
      }
      // B tile: transpose Wo[kc..kc+31][nb..nb+63] -> sB[n][k]
#pragma unroll
      for (int j = 0; j < 8; ++j) {
        int f = tid + 256 * j;
        int k = f >> 6, n = f & 63;
        sB[n * 32 + k] = Wob[(size_t)(kc + k) * EMB + nb + n];
      }
    }
    async_wait_all();
    __syncthreads();

    ABFrag a;
    {
      const uint4* p = (const uint4*)&sA[(wv * 16 + lm) * 32 + hi * 8];
      a.q[0] = p[0]; a.q[1] = p[2];
    }
#pragma unroll
    for (int t = 0; t < 4; ++t) {
      ABFrag bf;
      const uint4* p = (const uint4*)&sB[(t * 16 + lm) * 32 + hi * 8];
      bf.q[0] = p[0]; bf.q[1] = p[2];
      c[t] = __builtin_amdgcn_wmma_f32_16x16x32_bf16(
                 false, a.v, false, bf.v, (short)0, c[t], false, false);
    }
  }

#pragma unroll
  for (int t = 0; t < 4; ++t)
#pragma unroll
    for (int r = 0; r < 8; ++r) {
      int rowg = rb + wv * 16 + r + 8 * hi;
      out[(size_t)rowg * EMB + nb + t * 16 + lm] = c[t][r];
    }
}

// ---------------------------------------------------------------------------
extern "C" void kernel_launch(void* const* d_in, const int* in_sizes, int n_in,
                              void* d_out, int out_size, void* d_ws, size_t ws_size,
                              hipStream_t stream) {
  const float* x  = (const float*)d_in[0];
  const float* Wq = (const float*)d_in[1];
  const float* Wk = (const float*)d_in[2];
  const float* Wv = (const float*)d_in[3];
  const float* bq = (const float*)d_in[4];
  const float* bk = (const float*)d_in[5];
  const float* bv = (const float*)d_in[6];
  const float* Wo = (const float*)d_in[7];
  const float* bo = (const float*)d_in[8];
  float* out = (float*)d_out;

  // workspace layout (bf16 as ushort)
  const size_t BHSD = (size_t)BATCH * NH * SEQ * DH;  // 4,194,304
  unsigned short* Qb    = (unsigned short*)d_ws;
  unsigned short* Kb    = Qb + BHSD;
  unsigned short* Vtb   = Kb + BHSD;
  unsigned short* valsb = Vtb + BHSD;
  unsigned short* Wob   = valsb + (size_t)BATCH * SEQ * EMB;
  // total: (4*4,194,304 + 1,048,576) * 2 bytes = 34 MiB

  mha_cvt_kernel<<<512, 256, 0, stream>>>(Wo, Wob, EMB * EMB);

  mha_qkv_kernel<<<dim3(BATCH * NH, SEQ / 64), 128, 0, stream>>>(
      x, Wq, Wk, Wv, bq, bk, bv, Qb, Kb, Vtb);

  mha_attn_kernel<<<dim3(BATCH * NH, SEQ / 64), 128, 0, stream>>>(
      Qb, Kb, Vtb, valsb);

  mha_oproj_kernel<<<dim3(BATCH * SEQ / 128, EMB / 64), 256, 0, stream>>>(
      valsb, Wob, bo, out);
}
